// ModulatedConv_53936199303191
// MI455X (gfx1250) — matrix operand order repaired and checked by reading, not compiled, not run on hardware
//
#include <hip/hip_runtime.h>

// ---------------- problem constants ----------------
#define BB  8
#define CI  512
#define CO  512
#define HH  32
#define WW  32
#define P   1024          // HH*WW
#define OH  65            // pre-FIR spatial (2H+1)
#define GAIN   (1.0f/1536.0f)        // 1/sqrt(CO*CI*9)
#define GAIN2  (1.0f/2359296.0f)
#define EPSV   1e-8f

typedef __bf16 bf16_t;
typedef bf16_t v16bf __attribute__((ext_vector_type(16)));
typedef bf16_t v8bf  __attribute__((ext_vector_type(8)));
typedef float  v8f   __attribute__((ext_vector_type(8)));

static __device__ __forceinline__ v16bf cat16(v8bf a, v8bf b) {
  return __builtin_shufflevector(a, b, 0,1,2,3,4,5,6,7,8,9,10,11,12,13,14,15);
}

// ---- prep 1: wsq[o,i] = sum_kk w[o,i,kh,kw]^2 ----
__global__ void k_wsq(const float* __restrict__ w, float* __restrict__ wsq) {
  int t = blockIdx.x * 256 + threadIdx.x;
  if (t >= CO * CI) return;
  const float* p = w + (size_t)t * 9;
  float a = 0.f;
#pragma unroll
  for (int k = 0; k < 9; ++k) { float v = p[k]; a += v * v; }
  wsq[t] = a;
}

// ---- prep 2: dcoef[b,o] = rsqrt(gain^2 * sum_i (s+1)^2 wsq[o,i] + eps) ----
__global__ void k_dcoef(const float* __restrict__ styles,
                        const float* __restrict__ wsq,
                        float* __restrict__ dcoef) {
  __shared__ float red[256];
  int b = blockIdx.x >> 9, o = blockIdx.x & (CO - 1);
  float a = 0.f;
  for (int i = threadIdx.x; i < CI; i += 256) {
    float s = styles[b * CI + i] + 1.0f;
    a += s * s * wsq[o * CI + i];
  }
  red[threadIdx.x] = a;
  __syncthreads();
  for (int s2 = 128; s2 > 0; s2 >>= 1) {
    if (threadIdx.x < s2) red[threadIdx.x] += red[threadIdx.x + s2];
    __syncthreads();
  }
  if (threadIdx.x == 0) dcoef[blockIdx.x] = rsqrtf(red[0] * GAIN2 + EPSV);
}

// ---- prep 3: w * gain  ->  hi/lo bf16, layout [tap][o][i] ----
__global__ void k_wcvt(const float* __restrict__ w,
                       bf16_t* __restrict__ whi, bf16_t* __restrict__ wlo) {
  int tid = blockIdx.x * 256 + threadIdx.x;     // tid = (t<<18) | (o<<9) | i
  if (tid >= 9 * CO * CI) return;
  int i = tid & (CI - 1);
  int o = (tid >> 9) & (CO - 1);
  int t = tid >> 18;
  float v = w[((size_t)o * CI + i) * 9 + t] * GAIN;
  bf16_t h = (bf16_t)v;
  whi[tid] = h;
  wlo[tid] = (bf16_t)(v - (float)h);
}

// ---- prep 4: x * (s+1) -> hi/lo bf16, channel-last layout [b][pixel][i] ----
__global__ void k_xcvt(const float* __restrict__ x, const float* __restrict__ styles,
                       bf16_t* __restrict__ xhi, bf16_t* __restrict__ xlo) {
  int tid = blockIdx.x * 256 + threadIdx.x;     // tid = (b<<19) | (p<<9) | i
  if (tid >= BB * P * CI) return;
  int i = tid & (CI - 1);
  int p = (tid >> 9) & (P - 1);
  int b = tid >> 19;
  float s = styles[b * CI + i] + 1.0f;
  float v = x[((size_t)(b * CI + i)) * P + p] * s;
  bf16_t h = (bf16_t)v;
  xhi[tid] = h;
  xlo[tid] = (bf16_t)(v - (float)h);
}

// ---- main implicit GEMM per output-parity class, split-bf16 WMMA ----
// B panel (64 pixels) staged once per block in LDS via async global->LDS
// copies (double-buffered); 8 waves (M=128) consume it. Boundary pixels are
// sourced from a zeroed pad page -> no divergence in the K loop.
__global__ __launch_bounds__(256) void k_gemm(
    const bf16_t* __restrict__ whi, const bf16_t* __restrict__ wlo,
    const bf16_t* __restrict__ xhi, const bf16_t* __restrict__ xlo,
    const bf16_t* __restrict__ zpad,
    const float* __restrict__ dcoef, float* __restrict__ yacc,
    int py, int px, int GW, int GHW) {
  // [buf(2)][arr(2: hi,lo)][pixel(64)][40 bf16 = 32 K-vals + 8 pad]  => 20 KB
  __shared__ alignas(16) bf16_t smem[2 * 2 * 64 * 40];

  const int tid  = threadIdx.x;
  const int lane = tid & 31;
  const int wave = tid >> 5;
  const int hs   = lane >> 4;        // K half-select per ISA operand layout
  const int r    = lane & 15;        // A row / B column within 16x16 tile
  const int b    = blockIdx.z;
  const int obase = (blockIdx.y * 8 + wave) * 16;
  const int nbase = blockIdx.x * 64;

  // async-copy slots: 512 x 16B per chunk, 2 per thread
  const int arr0 = tid >> 8,         p0 = (tid >> 2) & 63,         sl0 = tid & 3;
  const int arr1 = (tid + 256) >> 8, p1 = ((tid + 256) >> 2) & 63, sl1 = (tid + 256) & 3;
  const uint32_t ldsbase = (uint32_t)(size_t)(void*)smem;
  const uint32_t dst0 = ldsbase + (uint32_t)(arr0 * 64 + p0) * 80u + (uint32_t)sl0 * 16u;
  const uint32_t dst1 = ldsbase + (uint32_t)(arr1 * 64 + p1) * 80u + (uint32_t)sl1 * 16u;

  const int n0 = nbase + p0, n1 = nbase + p1;
  const bool in0 = n0 < GHW, in1 = n1 < GHW;
  const int mm0 = (in0 ? n0 : 0) / GW, cc0 = (in0 ? n0 : 0) % GW;
  const int mm1 = (in1 ? n1 : 0) / GW, cc1 = (in1 ? n1 : 0) % GW;

  // per-lane wmma N-tile pixel info (LDS reads + epilogue)
  int nm[4], nc[4]; bool nin[4];
#pragma unroll
  for (int nt = 0; nt < 4; ++nt) {
    int n = nbase + nt * 16 + r;
    nin[nt] = n < GHW;
    int nn = nin[nt] ? n : 0;
    nm[nt] = nn / GW; nc[nt] = nn % GW;
  }

  v8f acc[4] = {};

  for (int kh = py; kh < 3; kh += 2) {          // taps with kh parity == py
    const int dy = (kh - py) >> 1;
    for (int kw = px; kw < 3; kw += 2) {
      const int dx = (kw - px) >> 1;
      const int t = kh * 3 + kw;

      // A row pointers (lane r -> row obase+r; K chunks per ISA A layout)
      const bf16_t* Ah = whi + (((size_t)t * CO + obase + r) << 9) + hs * 8;
      const bf16_t* Al = wlo + (((size_t)t * CO + obase + r) << 9) + hs * 8;

      // B copy sources for this tap (invalid -> zero pad page)
      const int iy0 = mm0 - dy, ix0 = cc0 - dx;
      const int iy1 = mm1 - dy, ix1 = cc1 - dx;
      const bool v0 = in0 && iy0 >= 0 && iy0 < HH && ix0 >= 0 && ix0 < WW;
      const bool v1 = in1 && iy1 >= 0 && iy1 < HH && ix1 >= 0 && ix1 < WW;
      const bf16_t* s0 = v0 ? (arr0 ? xlo : xhi) +
                              (((size_t)b * P + iy0 * WW + ix0) << 9) + sl0 * 8
                            : zpad + sl0 * 8;
      const bf16_t* s1 = v1 ? (arr1 ? xlo : xhi) +
                              (((size_t)b * P + iy1 * WW + ix1) << 9) + sl1 * 8
                            : zpad + sl1 * 8;

      // prime: chunk 0 -> buf 0
      asm volatile("global_load_async_to_lds_b128 %0, %1, off"
                   :: "v"(dst0), "v"(s0) : "memory");
      asm volatile("global_load_async_to_lds_b128 %0, %1, off"
                   :: "v"(dst1), "v"(s1) : "memory");

      for (int kk = 0; kk < 16; ++kk) {
        const int k0 = kk << 5;
        if (kk < 15) {                          // copy chunk kk+1 -> other buf
          const uint32_t bo = ((kk + 1) & 1) ? 10240u : 0u;
          asm volatile("global_load_async_to_lds_b128 %0, %1, off"
                       :: "v"(dst0 + bo), "v"(s0 + k0 + 32) : "memory");
          asm volatile("global_load_async_to_lds_b128 %0, %1, off"
                       :: "v"(dst1 + bo), "v"(s1 + k0 + 32) : "memory");
          asm volatile("s_wait_asynccnt 0x2" ::: "memory");   // chunk kk done
        } else {
          asm volatile("s_wait_asynccnt 0x0" ::: "memory");
        }
        __syncthreads();

        v16bf ah = cat16(*(const v8bf*)(Ah + k0), *(const v8bf*)(Ah + k0 + 16));
        v16bf al = cat16(*(const v8bf*)(Al + k0), *(const v8bf*)(Al + k0 + 16));
        const bf16_t* sb = smem + (size_t)(kk & 1) * 5120;    // buf select
#pragma unroll
        for (int nt = 0; nt < 4; ++nt) {
          const bf16_t* pb = sb + (size_t)(nt * 16 + r) * 40 + hs * 8;
          v16bf bh = cat16(*(const v8bf*)(pb), *(const v8bf*)(pb + 16));
          const bf16_t* pl = pb + 64 * 40;
          v16bf bl = cat16(*(const v8bf*)(pl), *(const v8bf*)(pl + 16));
          // split-bf16: hi*hi + hi*lo + lo*hi  (~f32 accuracy)
          acc[nt] = __builtin_amdgcn_wmma_f32_16x16x32_bf16(
              false, ah, false, bh, (short)0, acc[nt], false, false);
          acc[nt] = __builtin_amdgcn_wmma_f32_16x16x32_bf16(
              false, ah, false, bl, (short)0, acc[nt], false, false);
          acc[nt] = __builtin_amdgcn_wmma_f32_16x16x32_bf16(
              false, al, false, bh, (short)0, acc[nt], false, false);
        }
        __syncthreads();   // compute(kk) done before buf is overwritten
      }
    }
  }

  // epilogue: scale by dcoef[b,o], scatter into 65x65 grid (each pos once)
#pragma unroll
  for (int nt = 0; nt < 4; ++nt) {
    if (!nin[nt]) continue;
    int oy = 2 * nm[nt] + py, ox = 2 * nc[nt] + px;
#pragma unroll
    for (int j = 0; j < 8; ++j) {              // C/D layout: M = j + 8*hs, N = r
      int o = obase + hs * 8 + j;
      yacc[(((size_t)b * CO + o) * OH + oy) * OH + ox] =
          acc[nt][j] * dcoef[b * CO + o];
    }
  }
}

// ---- depthwise 4x4 FIR (pad 1), LDS-tiled: 65x65 -> 64x64 ----
__global__ __launch_bounds__(256) void k_fir(const float* __restrict__ yacc,
                                             float* __restrict__ out) {
  __shared__ float s[19][20];
  const int ch = blockIdx.z;                   // b*CO + o
  const int u0 = blockIdx.y * 16, v0 = blockIdx.x * 16;
  const size_t ybase = (size_t)ch * OH * OH;
  for (int li = threadIdx.x; li < 19 * 19; li += 256) {
    int rr = li / 19, cc = li % 19;
    int yy = u0 - 1 + rr, xx = v0 - 1 + cc;
    float v = 0.f;
    if (yy >= 0 && yy < OH && xx >= 0 && xx < OH) v = yacc[ybase + yy * OH + xx];
    s[rr][cc] = v;
  }
  __syncthreads();
  int ty = threadIdx.x >> 4, tx = threadIdx.x & 15;
  const float f1[4] = {1.f, 3.f, 3.f, 1.f};
  float a = 0.f;
#pragma unroll
  for (int fy = 0; fy < 4; ++fy)
#pragma unroll
    for (int fx = 0; fx < 4; ++fx)
      a += (f1[fy] * f1[fx]) * s[ty + fy][tx + fx];
  out[(size_t)ch * 4096 + (u0 + ty) * 64 + (v0 + tx)] = a * (1.0f / 16.0f);
}

extern "C" void kernel_launch(void* const* d_in, const int* in_sizes, int n_in,
                              void* d_out, int out_size, void* d_ws, size_t ws_size,
                              hipStream_t stream) {
  (void)in_sizes; (void)n_in; (void)out_size; (void)ws_size;
  const float* x      = (const float*)d_in[0];
  const float* styles = (const float*)d_in[1];
  const float* w      = (const float*)d_in[2];
  float* out = (float*)d_out;

  char* ws = (char*)d_ws;
  size_t off = 0;
  auto alloc = [&](size_t bytes) {
    void* p = ws + off;
    off += (bytes + 255) & ~(size_t)255;
    return p;
  };
  float*  wsq   = (float*) alloc((size_t)CO * CI * 4);
  float*  dcoef = (float*) alloc((size_t)BB * CO * 4);
  bf16_t* whi   = (bf16_t*)alloc((size_t)9 * CO * CI * 2);
  bf16_t* wlo   = (bf16_t*)alloc((size_t)9 * CO * CI * 2);
  bf16_t* xhi   = (bf16_t*)alloc((size_t)BB * P * CI * 2);
  bf16_t* xlo   = (bf16_t*)alloc((size_t)BB * P * CI * 2);
  bf16_t* zpad  = (bf16_t*)alloc(2048);
  float*  yacc  = (float*) alloc((size_t)BB * CO * OH * OH * 4);

  hipMemsetAsync(zpad, 0, 2048, stream);       // zero pad page (capture-safe)

  k_wsq  <<<(CO * CI + 255) / 256, 256, 0, stream>>>(w, wsq);
  k_dcoef<<<BB * CO,               256, 0, stream>>>(styles, wsq, dcoef);
  k_wcvt <<<(9 * CO * CI + 255) / 256, 256, 0, stream>>>(w, whi, wlo);
  k_xcvt <<<(BB * P * CI + 255) / 256, 256, 0, stream>>>(x, styles, xhi, xlo);

  for (int py = 0; py < 2; ++py)
    for (int px = 0; px < 2; ++px) {
      int GH = py ? 32 : 33, GW = px ? 32 : 33, GHW = GH * GW;
      dim3 g((GHW + 63) / 64, CO / 128, BB);
      k_gemm<<<g, 256, 0, stream>>>(whi, wlo, xhi, xlo, zpad, dcoef, yacc,
                                    py, px, GW, GHW);
    }

  k_fir<<<dim3(4, 4, BB * CO), 256, 0, stream>>>(yacc, out);
}
